// CEAlignment_35639638622648
// MI455X (gfx1250) — matrix operand order repaired and checked by reading, not compiled, not run on hardware
//
#include <hip/hip_runtime.h>
#include <hip/hip_bf16.h>
#include <math.h>

// ---------------------------------------------------------------------------
// CDNA5 (gfx1250) implementation of the cross-entropy alignment reference.
//  - 8 MLP GEMMs + the qq^T similarity run on v_wmma_f32_16x16x32_bf16
//  - f32 -> bf16 conversion happens during LDS staging (weights stay f32 in HBM)
//  - global->register loads are software-pipelined one K-tile ahead so the
//    HBM weight stream overlaps the WMMA pipe; global_prefetch primes GL2
//  - Sinkhorn runs as one persistent 512-thread workgroup per head (L2-resident)
// ---------------------------------------------------------------------------

typedef __attribute__((ext_vector_type(16))) __bf16 v16bf;
typedef __attribute__((ext_vector_type(8)))  __bf16 v8bf;
typedef __attribute__((ext_vector_type(8)))  float  v8f;

#define BM 64
#define BN 128
#define BK 32
#define APAD 8   // As row stride = BK+APAD = 40 bf16 = 80B (16B aligned)
#define BPAD 8   // Bs row stride = BN+BPAD = 136 bf16 = 272B (16B aligned)

__device__ __forceinline__ v8bf cvt8(float4 a, float4 b) {
    v8bf v;
    v[0] = (__bf16)a.x; v[1] = (__bf16)a.y; v[2] = (__bf16)a.z; v[3] = (__bf16)a.w;
    v[4] = (__bf16)b.x; v[5] = (__bf16)b.y; v[6] = (__bf16)b.z; v[7] = (__bf16)b.w;
    return v;
}

__device__ __forceinline__ v16bf load_frag(const __bf16* p0, const __bf16* p1) {
    v8bf lo = *(const v8bf*)p0;
    v8bf hi = *(const v8bf*)p1;
    return __builtin_shufflevector(lo, hi, 0,1,2,3,4,5,6,7,8,9,10,11,12,13,14,15);
}

__device__ __forceinline__ v8f wmma_bf16(v16bf a, v16bf b, v8f c) {
    // (neg_a, A, neg_b, B, c_mod, C, reuse_a, reuse_b)
    return __builtin_amdgcn_wmma_f32_16x16x32_bf16(false, a, false, b, (short)0, c,
                                                   false, false);
}

// ---------------------------------------------------------------------------
// GEMM: C[M x N] = act( A[M x K] @ B[K x N] + bias ), A/B f32 row-major.
// Block tile 64x128, BK=32. 256 threads = 8 waves, each wave owns 32x32.
// Global loads for tile k+1 are issued before the WMMA work of tile k.
// ---------------------------------------------------------------------------
__global__ __launch_bounds__(256) void gemm_bf16_wmma(
    const float* __restrict__ A, int lda,
    const float* __restrict__ B, int ldb,
    const float* __restrict__ bias,
    float* __restrict__ C, int ldc,
    int K, int relu)
{
    __shared__ __bf16 As[BM][BK + APAD];
    __shared__ __bf16 Bs[BK][BN + BPAD];

    const int t    = threadIdx.x;
    const int lane = t & 31;
    const int w    = t >> 5;
    const int wm   = (w & 1) * 32;   // wave tile row offset in block tile
    const int wn   = (w >> 1) * 32;  // wave tile col offset in block tile
    const int m0   = blockIdx.y * BM;
    const int n0   = blockIdx.x * BN;

    v8f acc[2][2] = {};

    // staging assignments
    const int ar = t >> 2, ac = (t & 3) * 8;   // A: 64 rows x 32 cols, 8 f32/thread
    const int br = t >> 3, bc = (t & 7) * 16;  // B: 32 rows x 128 cols, 16 f32/thread

    const int half = lane >> 4;
    const int r16  = lane & 15;

    const float* ag = A + (size_t)(m0 + ar) * lda + ac;
    const float* bg = B + (size_t)br * ldb + n0 + bc;

    // preload K-tile 0
    float4 ra0 = ((const float4*)ag)[0];
    float4 ra1 = ((const float4*)ag)[1];
    float4 rb0 = ((const float4*)bg)[0];
    float4 rb1 = ((const float4*)bg)[1];
    float4 rb2 = ((const float4*)bg)[2];
    float4 rb3 = ((const float4*)bg)[3];

    for (int k0 = 0; k0 < K; k0 += BK) {
        // ---- commit current tile to LDS (f32 -> bf16) ----
        *(v8bf*)&As[ar][ac]     = cvt8(ra0, ra1);
        *(v8bf*)&Bs[br][bc]     = cvt8(rb0, rb1);
        *(v8bf*)&Bs[br][bc + 8] = cvt8(rb2, rb3);
        __syncthreads();

        // ---- issue next tile's global loads; they fly under the WMMAs ----
        if (k0 + BK < K) {
            ag += BK;
            bg += (size_t)BK * ldb;
            ra0 = ((const float4*)ag)[0];
            ra1 = ((const float4*)ag)[1];
            rb0 = ((const float4*)bg)[0];
            rb1 = ((const float4*)bg)[1];
            rb2 = ((const float4*)bg)[2];
            rb3 = ((const float4*)bg)[3];
            // prime GL2 one further tile down the streamed weight matrix
            __builtin_prefetch(bg + (size_t)BK * ldb, 0, 1);
        }

        // ---- fragments from LDS per ISA 16-bit A(16x32)/B(32x16) layouts ----
        v16bf af[2], bfr[2];
#pragma unroll
        for (int i = 0; i < 2; ++i) {
            const __bf16* ap = &As[wm + i * 16 + r16][0];
            af[i] = load_frag(ap + half * 8, ap + 16 + half * 8);
        }
#pragma unroll
        for (int j = 0; j < 2; ++j) {
#pragma unroll
            for (int s = 0; s < 16; ++s)
                bfr[j][s] = Bs[(s & 7) + half * 8 + (s >> 3) * 16][wn + j * 16 + r16];
        }
#pragma unroll
        for (int i = 0; i < 2; ++i)
#pragma unroll
            for (int j = 0; j < 2; ++j)
                acc[i][j] = wmma_bf16(af[i], bfr[j], acc[i][j]);

        __syncthreads();
    }

    // ---- epilogue: bias (+ReLU), 16x16 f32 C layout: row=(lane/16)*8+r, col=lane%16
#pragma unroll
    for (int i = 0; i < 2; ++i)
#pragma unroll
        for (int j = 0; j < 2; ++j)
#pragma unroll
            for (int r = 0; r < 8; ++r) {
                int row = m0 + wm + i * 16 + half * 8 + r;
                int col = n0 + wn + j * 16 + r16;
                float v = acc[i][j][r] + bias[col];
                if (relu) v = fmaxf(v, 0.0f);
                C[(size_t)row * ldc + col] = v;
            }
}

// ---------------------------------------------------------------------------
// Head-normalize: q viewed as [512 rows][4 heads][512 embed]; per (row,head):
// (x - mean) / sqrt(var_unbiased + 1e-8), in place. One block per (row,head).
// ---------------------------------------------------------------------------
__global__ __launch_bounds__(256) void headnorm_kernel(float* __restrict__ q)
{
    const int row = blockIdx.x;                 // 0..2047 == b*4+h
    float* p = q + (size_t)(row >> 2) * 2048 + (row & 3) * 512;
    __shared__ float red[256];
    const int t = threadIdx.x;

    float x0 = p[t], x1 = p[t + 256];
    red[t] = x0 + x1;
    __syncthreads();
    for (int s = 128; s > 0; s >>= 1) {
        if (t < s) red[t] += red[t + s];
        __syncthreads();
    }
    float mean = red[0] * (1.0f / 512.0f);
    __syncthreads();

    float d0 = x0 - mean, d1 = x1 - mean;
    red[t] = d0 * d0 + d1 * d1;
    __syncthreads();
    for (int s = 128; s > 0; s >>= 1) {
        if (t < s) red[t] += red[t + s];
        __syncthreads();
    }
    float var = red[0] * (1.0f / 511.0f);
    float inv = 1.0f / sqrtf(var + 1e-8f);
    p[t]       = d0 * inv;
    p[t + 256] = d1 * inv;
}

// ---------------------------------------------------------------------------
// Similarity: per head h, Out_h[a][b] = exp( (q1n_a . q2n_b) / sqrt(512) ).
// Same WMMA pipeline; B tile is staged transposed (q2 is row-major over embed).
// ---------------------------------------------------------------------------
__global__ __launch_bounds__(256) void align_wmma(
    const float* __restrict__ Q1, const float* __restrict__ Q2,
    float* __restrict__ Out)
{
    const int h = blockIdx.z;
    const float* Aq = Q1 + h * 512;     // row stride 2048
    const float* Bq = Q2 + h * 512;     // row stride 2048
    float* O = Out + (size_t)h * 512 * 512;

    __shared__ __bf16 As[BM][BK + APAD];
    __shared__ __bf16 Bs[BK][BN + BPAD];

    const int t    = threadIdx.x;
    const int lane = t & 31;
    const int w    = t >> 5;
    const int wm   = (w & 1) * 32;
    const int wn   = (w >> 1) * 32;
    const int m0   = blockIdx.y * BM;
    const int n0   = blockIdx.x * BN;

    v8f acc[2][2] = {};

    const int ar = t >> 2, ac = (t & 3) * 8;  // A staging
    const int nl = t >> 1, kc = (t & 1) * 16; // B staging (transposed)

    const int half = lane >> 4;
    const int r16  = lane & 15;

    const float* ag = Aq + (size_t)(m0 + ar) * 2048 + ac;
    const float* bg = Bq + (size_t)(n0 + nl) * 2048 + kc;

    float4 ra0 = ((const float4*)ag)[0];
    float4 ra1 = ((const float4*)ag)[1];
    float4 rf0 = ((const float4*)bg)[0];
    float4 rf1 = ((const float4*)bg)[1];
    float4 rf2 = ((const float4*)bg)[2];
    float4 rf3 = ((const float4*)bg)[3];

    for (int k0 = 0; k0 < 512; k0 += BK) {
        *(v8bf*)&As[ar][ac] = cvt8(ra0, ra1);
        // scatter-transpose 16 embed values of one q2 row into Bs[k][n]
        {
            float vals[16] = {rf0.x, rf0.y, rf0.z, rf0.w, rf1.x, rf1.y, rf1.z, rf1.w,
                              rf2.x, rf2.y, rf2.z, rf2.w, rf3.x, rf3.y, rf3.z, rf3.w};
#pragma unroll
            for (int j = 0; j < 16; ++j) Bs[kc + j][nl] = (__bf16)vals[j];
        }
        __syncthreads();

        if (k0 + BK < 512) {
            ag += BK;
            bg += BK;
            ra0 = ((const float4*)ag)[0];
            ra1 = ((const float4*)ag)[1];
            rf0 = ((const float4*)bg)[0];
            rf1 = ((const float4*)bg)[1];
            rf2 = ((const float4*)bg)[2];
            rf3 = ((const float4*)bg)[3];
        }

        v16bf af[2], bfr[2];
#pragma unroll
        for (int i = 0; i < 2; ++i) {
            const __bf16* ap = &As[wm + i * 16 + r16][0];
            af[i] = load_frag(ap + half * 8, ap + 16 + half * 8);
        }
#pragma unroll
        for (int j = 0; j < 2; ++j) {
#pragma unroll
            for (int s = 0; s < 16; ++s)
                bfr[j][s] = Bs[(s & 7) + half * 8 + (s >> 3) * 16][wn + j * 16 + r16];
        }
#pragma unroll
        for (int i = 0; i < 2; ++i)
#pragma unroll
            for (int j = 0; j < 2; ++j)
                acc[i][j] = wmma_bf16(af[i], bfr[j], acc[i][j]);

        __syncthreads();
    }

    const float rscale = 0.04419417382415922f;  // 1/sqrt(512)
#pragma unroll
    for (int i = 0; i < 2; ++i)
#pragma unroll
        for (int j = 0; j < 2; ++j)
#pragma unroll
            for (int r = 0; r < 8; ++r) {
                int row = m0 + wm + i * 16 + half * 8 + r;
                int col = n0 + wn + j * 16 + r16;
                O[(size_t)row * 512 + col] = expf(acc[i][j][r] * rscale);
            }
}

// ---------------------------------------------------------------------------
// Sinkhorn: one 512-thread block per head, matrix (512x512) lives in L2.
// Faithful to reference: col-normalize -> row check -> row-normalize -> col
// check, early-exit keeps m1 when rows already match. Up to 500 iterations.
// ---------------------------------------------------------------------------
__global__ __launch_bounds__(512) void sinkhorn_kernel(
    float* __restrict__ W4,
    const float* __restrict__ p1g,  // [512][4]
    const float* __restrict__ p2g)  // [512][4]
{
    const int h = blockIdx.x;
    float* W = W4 + (size_t)h * 512 * 512;
    const int t = threadIdx.x;

    __shared__ float cs[512];
    __shared__ int okflag;

    const float p1t = p1g[t * 4 + h];   // this thread's row marginal
    const float p2t = p2g[t * 4 + h];   // this thread's col marginal
    float* row = W + (size_t)t * 512;

    for (int it = 0; it < 500; ++it) {
        // column sums (coalesced over lanes)
        float s = 0.0f;
        for (int a = 0; a < 512; ++a) s += W[(size_t)a * 512 + t];
        cs[t] = p2t / (s + 1e-8f);
        __syncthreads();

        // m1 = col-normalized; accumulate row sum on the fly
        float r = 0.0f;
        for (int b = 0; b < 512; ++b) {
            float v = row[b] * cs[b];
            row[b] = v;
            r += v;
        }
        int myok = fabsf(r - p1t) <= 0.01f;
        if (t == 0) okflag = 1;
        __syncthreads();
        if (!myok) okflag = 0;
        __syncthreads();
        if (okflag) break;              // rows match: keep m1, done

        // m2 = row-normalized
        float rf = p1t / (r + 1e-8f);
        for (int b = 0; b < 512; ++b) row[b] *= rf;
        __syncthreads();

        // column check on m2
        float s2 = 0.0f;
        for (int a = 0; a < 512; ++a) s2 += W[(size_t)a * 512 + t];
        int cok = fabsf(s2 - p2t) <= 0.01f;
        if (t == 0) okflag = 1;
        __syncthreads();
        if (!cok) okflag = 0;
        __syncthreads();
        if (okflag) break;              // cols match: keep m2, done
    }
}

// ---------------------------------------------------------------------------
// Permute [4][512][512] -> out[(a*512+b)*4 + h]  (coalesced writes)
// ---------------------------------------------------------------------------
__global__ __launch_bounds__(256) void permute_kernel(
    const float* __restrict__ W4, float* __restrict__ out)
{
    int o = blockIdx.x * 256 + threadIdx.x;     // 0 .. 4*512*512-1
    int h = o & 3;
    int b = (o >> 2) & 511;
    int a = o >> 11;
    out[o] = W4[(size_t)h * 262144 + (size_t)a * 512 + b];
}

// ---------------------------------------------------------------------------
extern "C" void kernel_launch(void* const* d_in, const int* in_sizes, int n_in,
                              void* d_out, int out_size, void* d_ws, size_t ws_size,
                              hipStream_t stream)
{
    (void)in_sizes; (void)n_in; (void)out_size; (void)ws_size;

    const float* x1  = (const float*)d_in[0];
    const float* x2  = (const float*)d_in[1];
    const float* x1p = (const float*)d_in[2];
    const float* x2p = (const float*)d_in[3];
    const float* w1[4], *b1[4], *w2[4], *b2[4];
    for (int i = 0; i < 4; ++i) {
        w1[i] = (const float*)d_in[4 + i];
        b1[i] = (const float*)d_in[8 + i];
        w2[i] = (const float*)d_in[12 + i];
        b2[i] = (const float*)d_in[16 + i];
    }

    float* ws = (float*)d_ws;
    const size_t ACT = (size_t)512 * 2048;       // 1M floats
    float* ping   = ws;
    float* pong   = ws + ACT;
    float* q1     = ws + 2 * ACT;
    float* q2     = ws + 3 * ACT;
    float* alignw = ws + 4 * ACT;                // 4 * 512 * 512 floats

    dim3 gemm_grid(2048 / BN, 512 / BM);         // (16, 8)
    dim3 blk(256);

    // MLP1: x1 -> q1
    gemm_bf16_wmma<<<gemm_grid, blk, 0, stream>>>(x1,   2048, w1[0], 2048, b1[0], ping, 2048, 2048, 1);
    gemm_bf16_wmma<<<gemm_grid, blk, 0, stream>>>(ping, 2048, w1[1], 2048, b1[1], pong, 2048, 2048, 1);
    gemm_bf16_wmma<<<gemm_grid, blk, 0, stream>>>(pong, 2048, w1[2], 2048, b1[2], ping, 2048, 2048, 1);
    gemm_bf16_wmma<<<gemm_grid, blk, 0, stream>>>(ping, 2048, w1[3], 2048, b1[3], q1,   2048, 2048, 0);

    // MLP2: x2 -> q2
    gemm_bf16_wmma<<<gemm_grid, blk, 0, stream>>>(x2,   2048, w2[0], 2048, b2[0], ping, 2048, 2048, 1);
    gemm_bf16_wmma<<<gemm_grid, blk, 0, stream>>>(ping, 2048, w2[1], 2048, b2[1], pong, 2048, 2048, 1);
    gemm_bf16_wmma<<<gemm_grid, blk, 0, stream>>>(pong, 2048, w2[2], 2048, b2[2], ping, 2048, 2048, 1);
    gemm_bf16_wmma<<<gemm_grid, blk, 0, stream>>>(ping, 2048, w2[3], 2048, b2[3], q2,   2048, 2048, 0);

    // per-head normalize (in place)
    headnorm_kernel<<<2048, blk, 0, stream>>>(q1);
    headnorm_kernel<<<2048, blk, 0, stream>>>(q2);

    // exp(q1n . q2n^T / sqrt(E)) per head
    align_wmma<<<dim3(512 / BN, 512 / BM, 4), blk, 0, stream>>>(q1, q2, alignw);

    // Sinkhorn per head
    sinkhorn_kernel<<<4, 512, 0, stream>>>(alignw, x1p, x2p);

    // [h][a][b] -> [(a*512+b)*4 + h]
    permute_kernel<<<4096, blk, 0, stream>>>(alignw, (float*)d_out);
}
